// deepGNN_45689862095587
// MI455X (gfx1250) — compile-verified, matrix operand order checked by reading, not run on hardware
//
#include <hip/hip_runtime.h>
#include <hip/hip_fp16.h>

// ---------------- problem constants (match reference) ----------------
#define NNODES 50000
#define NEDGES 800000
#define TSEQ   4
#define MROWS  (NNODES * TSEQ)   // 200000, multiple of 16

typedef _Float16 h16v __attribute__((ext_vector_type(16)));
typedef float    f8v  __attribute__((ext_vector_type(8)));

union FragU { h16v v; unsigned u[8]; };

__device__ __forceinline__ f8v wmma_f16(h16v a, h16v b, f8v c) {
  return __builtin_amdgcn_wmma_f32_16x16x32_f16(false, a, false, b, (short)0, c,
                                                false, false);
}

__device__ __forceinline__ void atom_add_f32(float* p, float v) {
  unsafeAtomicAdd(p, v);
}

__device__ __forceinline__ void load_frag(FragU& f, const _Float16* p) {
  const unsigned* u = (const unsigned*)p;
#pragma unroll
  for (int i = 0; i < 4; ++i) f.u[i] = u[i];
#pragma unroll
  for (int i = 0; i < 4; ++i) f.u[4 + i] = u[8 + i];
}

// ---------------------------------------------------------------------
// Generic WMMA GEMM:  C[M,Np] = act(A[M,Kp](f16) * Wt[Np,Kp]^T(f16) + bias)
// A row-major lda; Wt row-major ldw (W pre-transposed, K contiguous).
// One wave computes a 16x64 band (4 n-tiles) so each A fragment feeds 4
// WMMAs. M multiple of 16, Np multiple of 64, Kp multiple of 32.
// ACT: 0=none 1=relu 2=leakyrelu(0.01)
// ---------------------------------------------------------------------
template <int ACT, bool OUT16, bool OUT32>
__global__ void gemm_ws(const _Float16* __restrict__ A, int lda,
                        const _Float16* __restrict__ Wt, int ldw,
                        const float* __restrict__ bias,
                        float* __restrict__ C32, _Float16* __restrict__ C16,
                        int ldc, int M, int Np, int Kp) {
  const int lane    = threadIdx.x & 31;
  const int wave    = (blockIdx.x * blockDim.x + threadIdx.x) >> 5;
  const int nwaves  = (gridDim.x * blockDim.x) >> 5;
  const int mtiles  = M >> 4;
  const int ngroups = Np >> 6;          // groups of 4 n-tiles
  const int ksteps  = Kp >> 5;
  const int lm = lane & 15;
  const int kb = (lane >> 4) << 3;      // 0 or 8 (K base per ISA A/B layout)
  const int total = mtiles * ngroups;

  for (int t = wave; t < total; t += nwaves) {
    const int mt = t / ngroups;
    const int ng = t - mt * ngroups;
    const _Float16* Ap = A  + (size_t)(mt * 16 + lm) * lda + kb;
    const _Float16* Bp = Wt + (size_t)(ng * 64 + lm) * ldw + kb;
    f8v acc[4];
#pragma unroll
    for (int j = 0; j < 4; ++j) acc[j] = (f8v){0.f, 0.f, 0.f, 0.f, 0.f, 0.f, 0.f, 0.f};
    for (int ks = 0; ks < ksteps; ++ks) {
      FragU fa;
      load_frag(fa, Ap + ks * 32);
      __builtin_prefetch(Ap + ks * 32 + 32, 0, 1);
#pragma unroll
      for (int j = 0; j < 4; ++j) {
        FragU fb;
        load_frag(fb, Bp + (size_t)j * 16 * ldw + ks * 32);
        acc[j] = wmma_f16(fa.v, fb.v, acc[j]);
      }
    }
    const int mbase = mt * 16 + ((lane >> 4) << 3);
#pragma unroll
    for (int j = 0; j < 4; ++j) {
      const int   ncol = (ng * 4 + j) * 16 + lm;
      const float bb   = bias ? bias[ncol] : 0.f;
#pragma unroll
      for (int r = 0; r < 8; ++r) {
        float v = acc[j][r] + bb;
        if (ACT == 1) v = v > 0.f ? v : 0.f;
        if (ACT == 2) v = v > 0.f ? v : 0.01f * v;
        if (OUT32) C32[(size_t)(mbase + r) * ldc + ncol] = v;
        if (OUT16) C16[(size_t)(mbase + r) * ldc + ncol] = (_Float16)v;
      }
    }
  }
}

// ---------------------------------------------------------------------
// Fused GN edge kernel: 64 edges per 128-thread block (4 waves).
// Ae = [y[src](64) | y[dst](64) | eattr(3) | pad] (f16, Kp=160) in LDS
// H  = relu(Ae @ W1 + b1)  (f16 [64,256]) in LDS
// m  = H @ W2 + b2 ; atomically scattered into agg[dst]
// ---------------------------------------------------------------------
#define EB 64
__global__ void edge_msg_kernel(const _Float16* __restrict__ Y,      // [N,64] f16
                                const int* __restrict__ src,
                                const int* __restrict__ dst,
                                const float* __restrict__ eattr,     // [E,3]
                                const _Float16* __restrict__ W1t,    // [256,160]
                                const float* __restrict__ b1,        // [256]
                                const _Float16* __restrict__ W2t,    // [64,256]
                                const float* __restrict__ b2,        // [64]
                                float* __restrict__ agg) {           // [N,64]
  __shared__ _Float16 Ae[EB][160];
  __shared__ _Float16 Hs[EB][256];
  __shared__ int srcl[EB];
  __shared__ int dstl[EB];
  const int tid = threadIdx.x;             // 128 threads
  const int e0  = blockIdx.x * EB;

  if (tid < EB) {
    srcl[tid] = src[e0 + tid];
    dstl[tid] = dst[e0 + tid];
  }
  __syncthreads();

  // gather y[src], y[dst] (32 dwords per source each)
  for (int i = tid; i < EB * 32; i += 128) {
    const int row = i >> 5, c = i & 31;
    ((unsigned*)&Ae[row][0])[c]  = ((const unsigned*)(Y + (size_t)srcl[row] * 64))[c];
    ((unsigned*)&Ae[row][64])[c] = ((const unsigned*)(Y + (size_t)dstl[row] * 64))[c];
  }
  // edge attr + zero pad cols 128..159
  for (int i = tid; i < EB * 32; i += 128) {
    const int row = i >> 5, c = i & 31;
    float v = (c < 3) ? eattr[(size_t)(e0 + row) * 3 + c] : 0.f;
    Ae[row][128 + c] = (_Float16)v;
  }
  __syncthreads();

  const int lane = tid & 31, wave = tid >> 5;     // 4 waves, 16 rows each
  const int lm = lane & 15;
  const int kb = (lane >> 4) << 3;
  const int mb = wave * 16 + ((lane >> 4) << 3);

  // GEMM1: [16,160] x [160,256] -> relu -> Hs
  for (int nt = 0; nt < 16; ++nt) {
    f8v acc = {0.f, 0.f, 0.f, 0.f, 0.f, 0.f, 0.f, 0.f};
    const _Float16* Bp = W1t + (size_t)(nt * 16 + lm) * 160 + kb;
#pragma unroll
    for (int ks = 0; ks < 5; ++ks) {
      FragU fa, fb;
      load_frag(fa, &Ae[wave * 16 + lm][ks * 32 + kb]);
      load_frag(fb, Bp + ks * 32);
      acc = wmma_f16(fa.v, fb.v, acc);
    }
    const float bb = b1[nt * 16 + lm];
#pragma unroll
    for (int r = 0; r < 8; ++r) {
      float v = acc[r] + bb;
      v = v > 0.f ? v : 0.f;
      Hs[mb + r][nt * 16 + lm] = (_Float16)v;
    }
  }
  __syncthreads();

  // GEMM2: [16,256] x [256,64] -> atomic scatter to agg[dst]
  for (int nt = 0; nt < 4; ++nt) {
    f8v acc = {0.f, 0.f, 0.f, 0.f, 0.f, 0.f, 0.f, 0.f};
    const _Float16* Bp = W2t + (size_t)(nt * 16 + lm) * 256 + kb;
#pragma unroll
    for (int ks = 0; ks < 8; ++ks) {
      FragU fa, fb;
      load_frag(fa, &Hs[wave * 16 + lm][ks * 32 + kb]);
      load_frag(fb, Bp + ks * 32);
      acc = wmma_f16(fa.v, fb.v, acc);
    }
    const float bb = b2[nt * 16 + lm];
#pragma unroll
    for (int r = 0; r < 8; ++r) {
      float v = acc[r] + bb;
      atom_add_f32(&agg[(size_t)dstl[mb + r] * 64 + nt * 16 + lm], v);
    }
  }
}

// ---------------- helper VALU kernels ----------------
__global__ void cvt_wt_kernel(const float* __restrict__ W, _Float16* __restrict__ Wt,
                              int K, int N, int Kp, int total) {
  int i = blockIdx.x * blockDim.x + threadIdx.x;
  if (i >= total) return;
  int n = i / Kp, k = i - n * Kp;
  float v = (k < K && n < N) ? W[(size_t)k * N + n] : 0.f;
  Wt[i] = (_Float16)v;
}

__global__ void cvt_bias_kernel(const float* __restrict__ b, float* __restrict__ bp,
                                int n, int np) {
  int i = blockIdx.x * blockDim.x + threadIdx.x;
  if (i >= np) return;
  bp[i] = (i < n) ? b[i] : 0.f;
}

__global__ void zero_f32_kernel(float* __restrict__ p, int total) {
  int i = blockIdx.x * blockDim.x + threadIdx.x;
  if (i < total) p[i] = 0.f;
}

// y0 = x.reshape(N,T,2) @ up_w + up_b ; writes f32 + f16, zero pad 36..63
__global__ void up_kernel(const float* __restrict__ x, const float* __restrict__ upw,
                          const float* __restrict__ upb,
                          float* __restrict__ y32, _Float16* __restrict__ y16) {
  int m = blockIdx.x * blockDim.x + threadIdx.x;
  if (m >= MROWS) return;
  int n = m >> 2, t = m & 3;
  float x0 = x[n * 8 + t * 2], x1 = x[n * 8 + t * 2 + 1];
  size_t base = (size_t)m * 64;
  for (int j = 0; j < 64; ++j) {
    float v = (j < 36) ? (x0 * upw[j] + x1 * upw[36 + j] + upb[j]) : 0.f;
    y32[base + j] = v;
    y16[base + j] = (_Float16)v;
  }
}

// per (node, head) attention; T=4, DH=9
__global__ void attn_kernel(const _Float16* __restrict__ q, const _Float16* __restrict__ k,
                            const _Float16* __restrict__ v, _Float16* __restrict__ o) {
  int i = blockIdx.x * blockDim.x + threadIdx.x;
  if (i >= NNODES * 4) return;
  int n = i >> 2, h = i & 3;
  size_t base = (size_t)n * 4 * 64;
  int c0 = h * 9;
  float qh[4][9], kh[4][9], vh[4][9];
  for (int t = 0; t < 4; ++t)
    for (int d = 0; d < 9; ++d) {
      qh[t][d] = (float)q[base + t * 64 + c0 + d];
      kh[t][d] = (float)k[base + t * 64 + c0 + d];
      vh[t][d] = (float)v[base + t * 64 + c0 + d];
    }
  const float scale = 1.f / 3.f;   // 1/sqrt(9)
  for (int tq = 0; tq < 4; ++tq) {
    float s[4], mx = -1e30f;
    for (int tk = 0; tk < 4; ++tk) {
      float a = 0.f;
      for (int d = 0; d < 9; ++d) a += qh[tq][d] * kh[tk][d];
      s[tk] = a * scale;
      mx = fmaxf(mx, s[tk]);
    }
    float den = 0.f;
    for (int tk = 0; tk < 4; ++tk) { s[tk] = __expf(s[tk] - mx); den += s[tk]; }
    float inv = 1.f / den;
    for (int d = 0; d < 9; ++d) {
      float a = 0.f;
      for (int tk = 0; tk < 4; ++tk) a += s[tk] * vh[tk][d];
      o[base + tq * 64 + c0 + d] = (_Float16)(a * inv);
    }
  }
  if (h == 0)
    for (int t = 0; t < 4; ++t)
      for (int c = 36; c < 64; ++c) o[base + t * 64 + c] = (_Float16)0.f;
}

// y = LN(y + delta) over 36 dims (ld 64); refresh f16 copy
__global__ void addln36_kernel(float* __restrict__ y, const float* __restrict__ d,
                               const float* __restrict__ g, const float* __restrict__ b,
                               _Float16* __restrict__ y16) {
  int m = blockIdx.x * blockDim.x + threadIdx.x;
  if (m >= MROWS) return;
  size_t base = (size_t)m * 64;
  float tmp[36], mu = 0.f;
  for (int j = 0; j < 36; ++j) { float t = y[base + j] + d[base + j]; tmp[j] = t; mu += t; }
  mu *= (1.f / 36.f);
  float var = 0.f;
  for (int j = 0; j < 36; ++j) { float t = tmp[j] - mu; var += t * t; }
  var *= (1.f / 36.f);
  float inv = rsqrtf(var + 1e-5f);
  for (int j = 0; j < 36; ++j) {
    float o = (tmp[j] - mu) * inv * g[j] + b[j];
    y[base + j] = o;
    y16[base + j] = (_Float16)o;
  }
}

// y = LN(src) over 64 dims
__global__ void ln64_kernel(const float* __restrict__ s, const float* __restrict__ g,
                            const float* __restrict__ b, float* __restrict__ y,
                            _Float16* __restrict__ y16, int Mr) {
  int m = blockIdx.x * blockDim.x + threadIdx.x;
  if (m >= Mr) return;
  size_t base = (size_t)m * 64;
  float mu = 0.f;
  for (int j = 0; j < 64; ++j) mu += s[base + j];
  mu *= (1.f / 64.f);
  float var = 0.f;
  for (int j = 0; j < 64; ++j) { float t = s[base + j] - mu; var += t * t; }
  var *= (1.f / 64.f);
  float inv = rsqrtf(var + 1e-5f);
  for (int j = 0; j < 64; ++j) {
    float o = (s[base + j] - mu) * inv * g[j] + b[j];
    y[base + j] = o;
    y16[base + j] = (_Float16)o;
  }
}

// y += LN(upd) over 64 dims (GN residual)
__global__ void gn_addln_kernel(float* __restrict__ y, const float* __restrict__ u,
                                const float* __restrict__ g, const float* __restrict__ b,
                                _Float16* __restrict__ y16) {
  int m = blockIdx.x * blockDim.x + threadIdx.x;
  if (m >= NNODES) return;
  size_t base = (size_t)m * 64;
  float mu = 0.f;
  for (int j = 0; j < 64; ++j) mu += u[base + j];
  mu *= (1.f / 64.f);
  float var = 0.f;
  for (int j = 0; j < 64; ++j) { float t = u[base + j] - mu; var += t * t; }
  var *= (1.f / 64.f);
  float inv = rsqrtf(var + 1e-5f);
  for (int j = 0; j < 64; ++j) {
    float o = y[base + j] + ((u[base + j] - mu) * inv * g[j] + b[j]);
    y[base + j] = o;
    y16[base + j] = (_Float16)o;
  }
}

// pack [N, T*36] (from padded f16 rows) into [N,160] f16, zero pad 144..159
__global__ void repack_lin2_kernel(const _Float16* __restrict__ y16,
                                   _Float16* __restrict__ a2) {
  int i = blockIdx.x * blockDim.x + threadIdx.x;
  if (i >= NNODES * 160) return;
  int n = i / 160, j = i - n * 160;
  _Float16 v = (_Float16)0.f;
  if (j < 144) {
    int t = j / 36, c = j - t * 36;
    v = y16[(size_t)(n * 4 + t) * 64 + c];
  }
  a2[i] = v;
}

// pack [y(64) | agg(64)] -> [N,128] f16
__global__ void repack_up_kernel(const _Float16* __restrict__ y16,
                                 const float* __restrict__ agg,
                                 _Float16* __restrict__ a) {
  int i = blockIdx.x * blockDim.x + threadIdx.x;
  if (i >= NNODES * 128) return;
  int n = i / 128, j = i - n * 128;
  a[i] = (j < 64) ? y16[(size_t)n * 64 + j]
                  : (_Float16)agg[(size_t)n * 64 + (j - 64)];
}

// out = h @ dec_w2 + dec_b2  (128 -> 2)
__global__ void dec2_kernel(const _Float16* __restrict__ h, const float* __restrict__ w2,
                            const float* __restrict__ b2, float* __restrict__ out) {
  int n = blockIdx.x * blockDim.x + threadIdx.x;
  if (n >= NNODES) return;
  float a0 = b2[0], a1 = b2[1];
  for (int j = 0; j < 128; ++j) {
    float hv = (float)h[(size_t)n * 128 + j];
    a0 += hv * w2[j * 2 + 0];
    a1 += hv * w2[j * 2 + 1];
  }
  out[n * 2 + 0] = a0;
  out[n * 2 + 1] = a1;
}

// ---------------------------------------------------------------------
static inline int ceil_div(long long a, int b) { return (int)((a + b - 1) / b); }

extern "C" void kernel_launch(void* const* d_in, const int* in_sizes, int n_in,
                              void* d_out, int out_size, void* d_ws, size_t ws_size,
                              hipStream_t stream) {
  (void)in_sizes; (void)n_in; (void)out_size;
  const int Nn = NNODES, Ee = NEDGES, M = MROWS;

  // ---- inputs in setup_inputs() dict (insertion) order ----
  const float* x     = (const float*)d_in[0];
  const int*   ei    = (const int*)d_in[1];        // jnp int64 -> int32 (x64 off)
  const int*   src   = ei;
  const int*   dstp  = ei + Ee;
  const float* eattr = (const float*)d_in[2];
  int pi = 3;
  const float* up_w = (const float*)d_in[pi++];
  const float* up_b = (const float*)d_in[pi++];
  const float* tfp[6][16];
  for (int l = 0; l < 6; ++l)
    for (int j = 0; j < 16; ++j) tfp[l][j] = (const float*)d_in[pi++];
  const float* lin2_w = (const float*)d_in[pi++];
  const float* lin2_b = (const float*)d_in[pi++];
  const float* enc_ng = (const float*)d_in[pi++];
  const float* enc_nb = (const float*)d_in[pi++];
  const float* gnp[2][10];
  for (int g = 0; g < 2; ++g)
    for (int j = 0; j < 10; ++j) gnp[g][j] = (const float*)d_in[pi++];
  const float* dec_w1 = (const float*)d_in[pi++];
  const float* dec_b1 = (const float*)d_in[pi++];
  const float* dec_w2 = (const float*)d_in[pi++];
  const float* dec_b2 = (const float*)d_in[pi++];

  // ---- workspace carve ----
  char* w = (char*)d_ws;
  auto carve = [&](size_t bytes) {
    char* p = w;
    w += (bytes + 255) & ~(size_t)255;
    return p;
  };
  _Float16* wt_tf   = (_Float16*)carve((size_t)6 * 6 * 64 * 64 * 2);
  float*    bp_tf   = (float*)carve((size_t)6 * 6 * 64 * 4);
  _Float16* wt_lin2 = (_Float16*)carve((size_t)64 * 160 * 2);
  _Float16* wm1t[2]; _Float16* wm2t[2]; _Float16* wu1t[2]; _Float16* wu2t[2];
  for (int g = 0; g < 2; ++g) {
    wm1t[g] = (_Float16*)carve((size_t)256 * 160 * 2);
    wm2t[g] = (_Float16*)carve((size_t)64 * 256 * 2);
    wu1t[g] = (_Float16*)carve((size_t)256 * 128 * 2);
    wu2t[g] = (_Float16*)carve((size_t)64 * 256 * 2);
  }
  _Float16* wt_dec1 = (_Float16*)carve((size_t)128 * 64 * 2);

  float*    P0 = (float*)carve((size_t)M * 64 * 4);      // y f32 (residual)
  _Float16* P1 = (_Float16*)carve((size_t)M * 64 * 2);   // y f16
  _Float16* P2 = (_Float16*)carve((size_t)M * 64 * 2);   // q / lin2-A
  _Float16* P3 = (_Float16*)carve((size_t)M * 64 * 2);   // k / agg(f32)
  _Float16* P4 = (_Float16*)carve((size_t)M * 64 * 2);   // v / aup / hdec
  _Float16* P5 = (_Float16*)carve((size_t)M * 64 * 2);   // attn-out / ffn-h / hup
  float*    P6 = (float*)carve((size_t)M * 64 * 4);      // GEMM f32 outputs
  if ((size_t)(w - (char*)d_ws) > ws_size) return;       // scratch too small

  const int TB = 256;
  auto cvt = [&](const float* W, _Float16* Wt, int K, int Nc, int Kp, int Np) {
    int tot = Np * Kp;
    cvt_wt_kernel<<<ceil_div(tot, TB), TB, 0, stream>>>(W, Wt, K, Nc, Kp, tot);
  };

  // ---- weight conversions (tf: pad 36x36 -> 64x64; slots q,k,v,o,w1,w2) ----
  const int widx[6] = {0, 2, 4, 6, 10, 12};
  const int bidx[6] = {1, 3, 5, 7, 11, 13};
  for (int l = 0; l < 6; ++l)
    for (int j = 0; j < 6; ++j) {
      cvt(tfp[l][widx[j]], wt_tf + (size_t)(l * 6 + j) * 4096, 36, 36, 64, 64);
      cvt_bias_kernel<<<1, 64, 0, stream>>>(tfp[l][bidx[j]],
                                            bp_tf + (size_t)(l * 6 + j) * 64, 36, 64);
    }
  cvt(lin2_w, wt_lin2, 144, 64, 160, 64);
  for (int g = 0; g < 2; ++g) {
    cvt(gnp[g][0], wm1t[g], 131, 256, 160, 256);
    cvt(gnp[g][2], wm2t[g], 256, 64, 256, 64);
    cvt(gnp[g][4], wu1t[g], 128, 256, 128, 256);
    cvt(gnp[g][6], wu2t[g], 256, 64, 256, 64);
  }
  cvt(dec_w1, wt_dec1, 64, 128, 64, 128);

  // grid: one wave per 16x64 band
  auto gemm_blocks = [&](int Mr, int Np) { return ceil_div((Mr / 16) * (Np / 64), 8); };

  // ---- encoder ----
  up_kernel<<<ceil_div(M, TB), TB, 0, stream>>>(x, up_w, up_b, P0, P1);
  for (int l = 0; l < 6; ++l) {
    const _Float16* Wq = wt_tf + (size_t)(l * 6 + 0) * 4096;
    const _Float16* Wk = wt_tf + (size_t)(l * 6 + 1) * 4096;
    const _Float16* Wv = wt_tf + (size_t)(l * 6 + 2) * 4096;
    const _Float16* Wo = wt_tf + (size_t)(l * 6 + 3) * 4096;
    const _Float16* W1 = wt_tf + (size_t)(l * 6 + 4) * 4096;
    const _Float16* W2 = wt_tf + (size_t)(l * 6 + 5) * 4096;
    const float* Bq = bp_tf + (size_t)(l * 6 + 0) * 64;
    const float* Bk = bp_tf + (size_t)(l * 6 + 1) * 64;
    const float* Bv = bp_tf + (size_t)(l * 6 + 2) * 64;
    const float* Bo = bp_tf + (size_t)(l * 6 + 3) * 64;
    const float* B1 = bp_tf + (size_t)(l * 6 + 4) * 64;
    const float* B2 = bp_tf + (size_t)(l * 6 + 5) * 64;
    gemm_ws<0, true, false><<<gemm_blocks(M, 64), TB, 0, stream>>>(
        P1, 64, Wq, 64, Bq, nullptr, P2, 64, M, 64, 64);
    gemm_ws<0, true, false><<<gemm_blocks(M, 64), TB, 0, stream>>>(
        P1, 64, Wk, 64, Bk, nullptr, P3, 64, M, 64, 64);
    gemm_ws<0, true, false><<<gemm_blocks(M, 64), TB, 0, stream>>>(
        P1, 64, Wv, 64, Bv, nullptr, P4, 64, M, 64, 64);
    attn_kernel<<<ceil_div(Nn * 4, TB), TB, 0, stream>>>(P2, P3, P4, P5);
    gemm_ws<0, false, true><<<gemm_blocks(M, 64), TB, 0, stream>>>(
        P5, 64, Wo, 64, Bo, P6, nullptr, 64, M, 64, 64);
    addln36_kernel<<<ceil_div(M, TB), TB, 0, stream>>>(P0, P6, tfp[l][8], tfp[l][9], P1);
    gemm_ws<2, true, false><<<gemm_blocks(M, 64), TB, 0, stream>>>(
        P1, 64, W1, 64, B1, nullptr, P5, 64, M, 64, 64);
    gemm_ws<0, false, true><<<gemm_blocks(M, 64), TB, 0, stream>>>(
        P5, 64, W2, 64, B2, P6, nullptr, 64, M, 64, 64);
    addln36_kernel<<<ceil_div(M, TB), TB, 0, stream>>>(P0, P6, tfp[l][14], tfp[l][15], P1);
  }
  // lin2 + encoder LN  ->  node features [N,64] in P0 (f32) / P1 (f16)
  repack_lin2_kernel<<<ceil_div((long long)Nn * 160, TB), TB, 0, stream>>>(P1, P2);
  gemm_ws<0, false, true><<<gemm_blocks(Nn, 64), TB, 0, stream>>>(
      P2, 160, wt_lin2, 160, lin2_b, P6, nullptr, 64, Nn, 64, 160);
  ln64_kernel<<<ceil_div(Nn, TB), TB, 0, stream>>>(P6, enc_ng, enc_nb, P0, P1, Nn);

  // ---- GN layers ----
  for (int g = 0; g < 2; ++g) {
    float* agg = (float*)P3;
    zero_f32_kernel<<<ceil_div(Nn * 64, TB), TB, 0, stream>>>(agg, Nn * 64);
    edge_msg_kernel<<<Ee / EB, 128, 0, stream>>>(P1, src, dstp, eattr,
                                                 wm1t[g], gnp[g][1],
                                                 wm2t[g], gnp[g][3], agg);
    repack_up_kernel<<<ceil_div((long long)Nn * 128, TB), TB, 0, stream>>>(P1, agg, P4);
    gemm_ws<1, true, false><<<gemm_blocks(Nn, 256), TB, 0, stream>>>(
        P4, 128, wu1t[g], 128, gnp[g][5], nullptr, P5, 256, Nn, 256, 128);
    gemm_ws<0, false, true><<<gemm_blocks(Nn, 64), TB, 0, stream>>>(
        P5, 256, wu2t[g], 256, gnp[g][7], P6, nullptr, 64, Nn, 64, 256);
    gn_addln_kernel<<<ceil_div(Nn, TB), TB, 0, stream>>>(P0, P6, gnp[g][8], gnp[g][9], P1);
  }

  // ---- decoder ----
  gemm_ws<1, true, false><<<gemm_blocks(Nn, 128), TB, 0, stream>>>(
      P1, 64, wt_dec1, 64, dec_b1, nullptr, P4, 128, Nn, 128, 64);
  dec2_kernel<<<ceil_div(Nn, TB), TB, 0, stream>>>(P4, dec_w2, dec_b2, (float*)d_out);
}
